// DecoderLayer_31988916420693
// MI455X (gfx1250) — matrix operand order repaired and checked by reading, not compiled, not run on hardware
//
#include <hip/hip_runtime.h>
#include <hip/hip_bf16.h>

typedef __bf16 bf16_t;
typedef bf16_t v16bf __attribute__((ext_vector_type(16)));
typedef bf16_t v8bf  __attribute__((ext_vector_type(8)));
typedef float  v8f   __attribute__((ext_vector_type(8)));
typedef __attribute__((ext_vector_type(4))) int v4i;

#define EMBED 2048
#define NH    16
#define NKV   4
#define HD    128
#define FFDIM 5632
#define SEQ   2048
#define BATCH 2
#define MROWS (BATCH * SEQ)           /* 4096 */
#define QKVN  ((NH + 2 * NKV) * HD)   /* 3072 */

// ---------------- async global->LDS staging (CDNA5 path, guarded) ----------

#if __has_builtin(__builtin_amdgcn_global_load_async_to_lds_b128)
#define HAVE_ASYNC_LDS 1
#else
#define HAVE_ASYNC_LDS 0
#endif

typedef v4i __attribute__((address_space(1)))* g4_ptr;  // int4 __device__*
typedef v4i __attribute__((address_space(3)))* l4_ptr;  // int4 __shared__*

__device__ __forceinline__ void stage16(const bf16_t* g, bf16_t* l) {
#if HAVE_ASYNC_LDS
  g4_ptr gp = reinterpret_cast<g4_ptr>((unsigned long long)g);
  l4_ptr lp = reinterpret_cast<l4_ptr>((unsigned int)(unsigned long long)l);
  __builtin_amdgcn_global_load_async_to_lds_b128(gp, lp, 0, 0);
#else
  *(v8bf*)l = *(const v8bf*)g;
#endif
}

// allow one stage (5 async ops) to stay in flight
__device__ __forceinline__ void async_wait_grp() {
#if HAVE_ASYNC_LDS && __has_builtin(__builtin_amdgcn_s_wait_asynccnt)
  __builtin_amdgcn_s_wait_asynccnt(5);
#endif
}
__device__ __forceinline__ void async_wait_all() {
#if HAVE_ASYNC_LDS && __has_builtin(__builtin_amdgcn_s_wait_asynccnt)
  __builtin_amdgcn_s_wait_asynccnt(0);
#endif
}

// ---------------- fragment helpers (bf16 WMMA 16x16x32) ----------------

__device__ __forceinline__ v16bf make_frag(const bf16_t* p0, const bf16_t* p1) {
  v8bf lo = *(const v8bf*)p0;
  v8bf hi = *(const v8bf*)p1;
  return __builtin_shufflevector(lo, hi, 0, 1, 2, 3, 4, 5, 6, 7,
                                         8, 9, 10, 11, 12, 13, 14, 15);
}

// A-matrix 16xK tile from row-major A (stride lda).
__device__ __forceinline__ v16bf load_a_frag(const bf16_t* A, int lda,
                                             int row0, int k0, int lane) {
  int r = lane & 15, hi = lane >> 4;
  const bf16_t* base = A + (size_t)(row0 + r) * lda + k0 + hi * 16;
  return make_frag(base, base + 8);
}

// B-matrix KxN tile sourced from row-major W[N x K] (computing A @ W^T).
__device__ __forceinline__ v16bf load_b_frag(const bf16_t* W, int ldw,
                                             int n0, int k0, int lane) {
  int n = lane & 15, hi = lane >> 4;
  const bf16_t* base = W + (size_t)(n0 + n) * ldw + k0 + hi * 16;
  return make_frag(base, base + 8);
}

__device__ __forceinline__ v8f wmma_bf16(v16bf a, v16bf b, v8f c) {
  return __builtin_amdgcn_wmma_f32_16x16x32_bf16(false, a, false, b,
                                                 (short)0, c, false, false);
}

// ---------------- fp32 -> bf16 weight conversion ----------------

__global__ __launch_bounds__(256) void cvt_f32_bf16_kernel(
    const float* __restrict__ src, bf16_t* __restrict__ dst, int n4) {
  int i = (blockIdx.x * 256 + threadIdx.x) * 4;
  if (i < n4 * 4) {
    float4 v = *(const float4*)(src + i);
    dst[i + 0] = (bf16_t)v.x;
    dst[i + 1] = (bf16_t)v.y;
    dst[i + 2] = (bf16_t)v.z;
    dst[i + 3] = (bf16_t)v.w;
  }
}

// ---------------- RMSNorm (fp32 in, bf16 out) ----------------

__global__ __launch_bounds__(256) void rmsnorm_kernel(
    const float* __restrict__ X, const float* __restrict__ w,
    bf16_t* __restrict__ out) {
  int row = blockIdx.x;
  const float* x = X + (size_t)row * EMBED;
  float ss = 0.f;
  for (int i = threadIdx.x; i < EMBED; i += 256) {
    float v = x[i];
    ss += v * v;
  }
  __shared__ float red[256];
  red[threadIdx.x] = ss;
  __syncthreads();
  for (int s = 128; s > 0; s >>= 1) {
    if (threadIdx.x < s) red[threadIdx.x] += red[threadIdx.x + s];
    __syncthreads();
  }
  float rms = rsqrtf(red[0] * (1.0f / EMBED) + 1e-5f);
  for (int i = threadIdx.x; i < EMBED; i += 256)
    out[(size_t)row * EMBED + i] = (bf16_t)(x[i] * rms * w[i]);
}

// ---------------- GEMM: C = A(bf16,MxK) @ W(bf16,NxK)^T ----------------
// Block: 256 thr = 8 waves (2Mw x 4Nw) -> 64(M) x 256(N); wave 32x64.
// A/B slabs async-staged into double-buffered LDS (K step 32, unroll 64).
// Requires K % 64 == 0 (2048, 5632 both qualify).

template <int EPI>
__global__ __launch_bounds__(256) void gemm_bf16_kernel(
    const bf16_t* __restrict__ A, const bf16_t* __restrict__ W,
    const float* resid, float* outF, bf16_t* outB,
    int M, int N, int K) {
  __shared__ __align__(16) bf16_t As[2][64][40];
  __shared__ __align__(16) bf16_t Bs[2][256][40];

  const int lane = threadIdx.x & 31;
  const int wave = threadIdx.x >> 5;
  const int mw = wave >> 2, nw = wave & 3;
  const int mblk = blockIdx.x * 64, nblk = blockIdx.y * 256;

  v8f acc[2][4] = {};

  auto stage = [&](int buf, int k0) {
    int row = threadIdx.x >> 2, ko = (threadIdx.x & 3) * 8;
    stage16(A + (size_t)(mblk + row) * K + k0 + ko, &As[buf][row][ko]);
#pragma unroll
    for (int t = 0; t < 4; ++t) {
      int c = threadIdx.x + t * 256;
      int br = c >> 2, bko = (c & 3) * 8;
      stage16(W + (size_t)(nblk + br) * K + k0 + bko, &Bs[buf][br][bko]);
    }
  };
  auto compute = [&](int buf) {
    v16bf af[2];
#pragma unroll
    for (int i = 0; i < 2; ++i) {
      const bf16_t* p = &As[buf][mw * 32 + i * 16 + (lane & 15)][(lane >> 4) * 16];
      af[i] = make_frag(p, p + 8);
    }
#pragma unroll
    for (int j = 0; j < 4; ++j) {
      const bf16_t* p = &Bs[buf][nw * 64 + j * 16 + (lane & 15)][(lane >> 4) * 16];
      v16bf bfr = make_frag(p, p + 8);
#pragma unroll
      for (int i = 0; i < 2; ++i) acc[i][j] = wmma_bf16(af[i], bfr, acc[i][j]);
    }
  };

  stage(0, 0);
  for (int k0 = 0; k0 < K; k0 += 64) {
    stage(1, k0 + 32);          // K % 64 == 0 -> always in range
    async_wait_grp();           // buf0 slab complete (<=5 in flight)
    __syncthreads();
    compute(0);
    __syncthreads();
    if (k0 + 64 < K) {
      stage(0, k0 + 64);
      async_wait_grp();
    } else {
      async_wait_all();
    }
    __syncthreads();
    compute(1);
    __syncthreads();
  }

  const int n = lane & 15, mb = (lane >> 4) * 8;
#pragma unroll
  for (int i = 0; i < 2; ++i)
#pragma unroll
    for (int j = 0; j < 4; ++j)
#pragma unroll
      for (int e = 0; e < 8; ++e) {
        size_t idx = (size_t)(mblk + mw * 32 + i * 16 + mb + e) * N +
                     nblk + nw * 64 + j * 16 + n;
        if (EPI == 0)
          outB[idx] = (bf16_t)acc[i][j][e];
        else
          outF[idx] = acc[i][j][e] + resid[idx];
      }
}

// ---------------- RoPE on q,k inside packed QKV (bf16, in-place) -----------

__global__ __launch_bounds__(256) void rope_kernel(
    bf16_t* QKV, const float* __restrict__ freqs) {
  const int pairsPerRow = (NH + NKV) * (HD / 2);  // 1280
  int pid = blockIdx.x * 256 + threadIdx.x;
  if (pid >= MROWS * pairsPerRow) return;
  int row = pid / pairsPerRow;
  int p = pid - row * pairsPerRow;
  int head = p >> 6;
  int dp = p & 63;
  int col = (head < NH) ? head * HD + dp * 2
                        : NH * HD + (head - NH) * HD + dp * 2;
  int s = row & (SEQ - 1);
  float c = freqs[(s * 64 + dp) * 2 + 0];
  float sn = freqs[(s * 64 + dp) * 2 + 1];
  bf16_t* q = QKV + (size_t)row * QKVN + col;
  float x0 = (float)q[0], x1 = (float)q[1];
  q[0] = (bf16_t)(x0 * c - x1 * sn);
  q[1] = (bf16_t)(x1 * c + x0 * sn);
}

// ---------------- Flash attention (causal, GQA 4:1, bf16 WMMA) -------------
// grid (S/64, NH, B); block 128 = 4 waves; each wave owns 16 queries.

__global__ __launch_bounds__(128) void flash_attn_kernel(
    const bf16_t* __restrict__ QKV, bf16_t* __restrict__ Out) {
  const int b = blockIdx.z, h = blockIdx.y, qblk = blockIdx.x;
  const int kvh = h / (NH / NKV);
  const int lane = threadIdx.x & 31, wave = threadIdx.x >> 5;
  const int q0 = qblk * 64 + wave * 16;

  const bf16_t* Qb = QKV + (size_t)b * SEQ * QKVN + h * HD;
  const bf16_t* Kb = QKV + (size_t)b * SEQ * QKVN + NH * HD + kvh * HD;
  const bf16_t* Vb = QKV + (size_t)b * SEQ * QKVN + (NH + NKV) * HD + kvh * HD;

  __shared__ __align__(16) bf16_t Vt[HD][40];     // V chunk, transposed
  __shared__ __align__(16) bf16_t Pl[4][16][40];  // per-wave P tile

  v16bf qf[4];
#pragma unroll
  for (int kk = 0; kk < 4; ++kk) qf[kk] = load_a_frag(Qb, QKVN, q0, kk * 32, lane);

  v8f o[8] = {};
  float mrow[8], lrow[8];
#pragma unroll
  for (int e = 0; e < 8; ++e) { mrow[e] = -1e30f; lrow[e] = 0.f; }

  const float scale = 0.022097086912079608f;  // 1/sqrt(2048)
  const int n = lane & 15, mb = (lane >> 4) * 8, hi = lane >> 4;
  const int kv_end = qblk * 64 + 64;

  for (int kv0 = 0; kv0 < kv_end; kv0 += 32) {
    __syncthreads();
    for (int idx = threadIdx.x; idx < 32 * HD; idx += 128) {
      int kv = idx >> 7, d = idx & 127;
      Vt[d][kv] = Vb[(size_t)(kv0 + kv) * QKVN + d];
    }
    __syncthreads();

    if (kv0 > q0 + 15) continue;  // fully masked for this wave

    v8f st[2];
#pragma unroll
    for (int t = 0; t < 2; ++t) {
      v16bf bk[4];
#pragma unroll
      for (int kk = 0; kk < 4; ++kk)
        bk[kk] = load_b_frag(Kb, QKVN, kv0 + t * 16, kk * 32, lane);
      v8f s = {};
#pragma unroll
      for (int kk = 0; kk < 4; ++kk) s = wmma_bf16(qf[kk], bk[kk], s);
      int col = kv0 + t * 16 + n;
#pragma unroll
      for (int e = 0; e < 8; ++e) {
        int row = q0 + mb + e;
        float v = s[e] * scale;
        s[e] = (col <= row) ? v : -1e30f;
      }
      st[t] = s;
    }

    float alpha[8];
#pragma unroll
    for (int e = 0; e < 8; ++e) {
      float mx = fmaxf(st[0][e], st[1][e]);
#pragma unroll
      for (int off = 8; off > 0; off >>= 1) mx = fmaxf(mx, __shfl_xor(mx, off, 32));
      float mnew = fmaxf(mrow[e], mx);
      float p0 = __expf(st[0][e] - mnew);
      float p1 = __expf(st[1][e] - mnew);
      st[0][e] = p0;
      st[1][e] = p1;
      float ps = p0 + p1;
#pragma unroll
      for (int off = 8; off > 0; off >>= 1) ps += __shfl_xor(ps, off, 32);
      alpha[e] = __expf(mrow[e] - mnew);
      lrow[e] = lrow[e] * alpha[e] + ps;
      mrow[e] = mnew;
    }
#pragma unroll
    for (int j = 0; j < 8; ++j)
#pragma unroll
      for (int e = 0; e < 8; ++e) o[j][e] *= alpha[e];

#pragma unroll
    for (int t = 0; t < 2; ++t)
#pragma unroll
      for (int e = 0; e < 8; ++e)
        Pl[wave][mb + e][t * 16 + n] = (bf16_t)st[t][e];

    const bf16_t* Prow = &Pl[wave][lane & 15][0];
    v16bf pf = make_frag(Prow + hi * 16, Prow + hi * 16 + 8);
#pragma unroll
    for (int j = 0; j < 8; ++j) {
      const bf16_t* Vrow = &Vt[j * 16 + n][hi * 16];
      v16bf vf = make_frag(Vrow, Vrow + 8);
      o[j] = wmma_bf16(pf, vf, o[j]);
    }
  }

#pragma unroll
  for (int j = 0; j < 8; ++j)
#pragma unroll
    for (int e = 0; e < 8; ++e)
      Out[(size_t)(b * SEQ + q0 + mb + e) * EMBED + h * HD + j * 16 + n] =
          (bf16_t)(o[j][e] / lrow[e]);
}

// ---------------- fused FFN up: silu(G@W1^T) * (G@W2^T) -> bf16 ------------
// Block: 8 waves (2Mw x 4Nw) -> 64(M) x 128(N). Wave: 32(M) x 32(N) of u & t.
// G/W1/W2 slabs async-staged into double-buffered LDS.

__global__ __launch_bounds__(256) void ffn_fused_kernel(
    const bf16_t* __restrict__ G, const bf16_t* __restrict__ W1,
    const bf16_t* __restrict__ W2, bf16_t* __restrict__ outB) {
  __shared__ __align__(16) bf16_t As[2][64][40];
  __shared__ __align__(16) bf16_t B1s[2][128][40];
  __shared__ __align__(16) bf16_t B2s[2][128][40];

  const int lane = threadIdx.x & 31;
  const int wave = threadIdx.x >> 5;
  const int mw = wave >> 2, nw = wave & 3;
  const int mblk = blockIdx.x * 64, nblk = blockIdx.y * 128;

  v8f au[2][2] = {}, at[2][2] = {};

  auto stage = [&](int buf, int k0) {
    int row = threadIdx.x >> 2, ko = (threadIdx.x & 3) * 8;
    stage16(G + (size_t)(mblk + row) * EMBED + k0 + ko, &As[buf][row][ko]);
#pragma unroll
    for (int t = 0; t < 2; ++t) {
      int c = threadIdx.x + t * 256;
      int br = c >> 2, bko = (c & 3) * 8;
      stage16(W1 + (size_t)(nblk + br) * EMBED + k0 + bko, &B1s[buf][br][bko]);
      stage16(W2 + (size_t)(nblk + br) * EMBED + k0 + bko, &B2s[buf][br][bko]);
    }
  };
  auto compute = [&](int buf) {
    v16bf af[2];
#pragma unroll
    for (int i = 0; i < 2; ++i) {
      const bf16_t* p = &As[buf][mw * 32 + i * 16 + (lane & 15)][(lane >> 4) * 16];
      af[i] = make_frag(p, p + 8);
    }
#pragma unroll
    for (int j = 0; j < 2; ++j) {
      const bf16_t* p1 = &B1s[buf][nw * 32 + j * 16 + (lane & 15)][(lane >> 4) * 16];
      v16bf b1 = make_frag(p1, p1 + 8);
      const bf16_t* p2 = &B2s[buf][nw * 32 + j * 16 + (lane & 15)][(lane >> 4) * 16];
      v16bf b2 = make_frag(p2, p2 + 8);
#pragma unroll
      for (int i = 0; i < 2; ++i) {
        au[i][j] = wmma_bf16(af[i], b1, au[i][j]);
        at[i][j] = wmma_bf16(af[i], b2, at[i][j]);
      }
    }
  };

  stage(0, 0);
  for (int k0 = 0; k0 < EMBED; k0 += 64) {
    stage(1, k0 + 32);
    async_wait_grp();
    __syncthreads();
    compute(0);
    __syncthreads();
    if (k0 + 64 < EMBED) {
      stage(0, k0 + 64);
      async_wait_grp();
    } else {
      async_wait_all();
    }
    __syncthreads();
    compute(1);
    __syncthreads();
  }

  const int n = lane & 15, mb = (lane >> 4) * 8;
#pragma unroll
  for (int i = 0; i < 2; ++i)
#pragma unroll
    for (int j = 0; j < 2; ++j)
#pragma unroll
      for (int e = 0; e < 8; ++e) {
        float u = au[i][j][e], t = at[i][j][e];
        float sg = u / (1.f + __expf(-u));
        outB[(size_t)(mblk + mw * 32 + i * 16 + mb + e) * FFDIM +
             nblk + nw * 32 + j * 16 + n] = (bf16_t)(sg * t);
      }
}

// ---------------- launch ----------------

extern "C" void kernel_launch(void* const* d_in, const int* in_sizes, int n_in,
                              void* d_out, int out_size, void* d_ws, size_t ws_size,
                              hipStream_t stream) {
  const float* x       = (const float*)d_in[0];
  const float* freqs   = (const float*)d_in[2];
  const float* w_qkv   = (const float*)d_in[4];
  const float* w_fc    = (const float*)d_in[5];
  const float* w1      = (const float*)d_in[6];
  const float* w2      = (const float*)d_in[7];
  const float* w3      = (const float*)d_in[8];
  const float* attn_nw = (const float*)d_in[9];
  const float* ff_nw   = (const float*)d_in[10];
  float* out = (float*)d_out;

  char* wsp = (char*)d_ws;
  auto carve = [&](size_t bytes) {
    char* p = wsp;
    wsp += (bytes + 255) & ~(size_t)255;
    return p;
  };
  bf16_t* wqkv_b = (bf16_t*)carve((size_t)QKVN * EMBED * 2);
  bf16_t* wfc_b  = (bf16_t*)carve((size_t)EMBED * EMBED * 2);
  bf16_t* w1_b   = (bf16_t*)carve((size_t)FFDIM * EMBED * 2);
  bf16_t* w2_b   = (bf16_t*)carve((size_t)FFDIM * EMBED * 2);
  bf16_t* w3_b   = (bf16_t*)carve((size_t)EMBED * FFDIM * 2);
  bf16_t* h_b    = (bf16_t*)carve((size_t)MROWS * EMBED * 2);
  bf16_t* qkv_b  = (bf16_t*)carve((size_t)MROWS * QKVN * 2);
  bf16_t* attn_b = (bf16_t*)carve((size_t)MROWS * EMBED * 2);
  bf16_t* g_b    = (bf16_t*)carve((size_t)MROWS * EMBED * 2);
  bf16_t* ff_b   = (bf16_t*)carve((size_t)MROWS * FFDIM * 2);

  auto cvt = [&](const float* s, bf16_t* d, size_t nel) {
    int n4 = (int)(nel / 4);
    cvt_f32_bf16_kernel<<<(n4 + 255) / 256, 256, 0, stream>>>(s, d, n4);
  };
  cvt(w_qkv, wqkv_b, (size_t)QKVN * EMBED);
  cvt(w_fc, wfc_b, (size_t)EMBED * EMBED);
  cvt(w1, w1_b, (size_t)FFDIM * EMBED);
  cvt(w2, w2_b, (size_t)FFDIM * EMBED);
  cvt(w3, w3_b, (size_t)EMBED * FFDIM);

  // h = rmsnorm(x, attn_norm_w)
  rmsnorm_kernel<<<MROWS, 256, 0, stream>>>(x, attn_nw, h_b);

  // qkv = h @ w_qkv^T
  gemm_bf16_kernel<0><<<dim3(MROWS / 64, QKVN / 256), 256, 0, stream>>>(
      h_b, wqkv_b, nullptr, nullptr, qkv_b, MROWS, QKVN, EMBED);

  // rope on q,k
  {
    int total = MROWS * (NH + NKV) * (HD / 2);
    rope_kernel<<<(total + 255) / 256, 256, 0, stream>>>(qkv_b, freqs);
  }

  // attention -> attn_b (b, s, h*d)
  flash_attn_kernel<<<dim3(SEQ / 64, NH, BATCH), 128, 0, stream>>>(qkv_b, attn_b);

  // h2 = x + attn @ w_fc^T   (fp32, into d_out)
  gemm_bf16_kernel<1><<<dim3(MROWS / 64, EMBED / 256), 256, 0, stream>>>(
      attn_b, wfc_b, x, out, nullptr, MROWS, EMBED, EMBED);

  // g = rmsnorm(h2, ff_norm_w)
  rmsnorm_kernel<<<MROWS, 256, 0, stream>>>(out, ff_nw, g_b);

  // ff = silu(g@w1^T) * (g@w2^T)
  ffn_fused_kernel<<<dim3(MROWS / 64, FFDIM / 128), 256, 0, stream>>>(
      g_b, w1_b, w2_b, ff_b);

  // out = h2 + ff @ w3^T
  gemm_bf16_kernel<1><<<dim3(MROWS / 64, EMBED / 256), 256, 0, stream>>>(
      ff_b, w3_b, out, out, nullptr, MROWS, EMBED, FFDIM);
}
